// FineAMT_56238301774000
// MI455X (gfx1250) — matrix operand order repaired and checked by reading, not compiled, not run on hardware
//
#include <hip/hip_runtime.h>
#include <hip/hip_bf16.h>
#include <math.h>

// ---------------- model constants ----------------
#define BLOCKS   6
#define DIMD     768
#define D_STATE  16
#define D_CONV   4
#define HEADDIM  64
#define D_INNER  1536
#define NHEADS   24
#define CONV_DIMC 1568
#define D_IN_PROJ 3128
#define NPITCH   128
#define BATCH    4
#define SEQLEN   4096
#define MROWS    (BATCH*SEQLEN)   // 16384

// ---------------- types ----------------
typedef __attribute__((ext_vector_type(16))) __bf16 v16bf;
typedef __attribute__((ext_vector_type(8)))  float  v8f;
typedef __attribute__((ext_vector_type(4)))  unsigned int u32x4;
typedef __attribute__((ext_vector_type(8)))  int i32x8;
typedef __attribute__((ext_vector_type(4)))  int i32x4;

struct __align__(16) F4 { float x, y, z, w; };
struct __align__(16) U4 { unsigned int x, y, z, w; };
union Frag { v16bf v; U4 q[2]; };

__device__ inline unsigned short f2bf(float f) {
    unsigned int u = __builtin_bit_cast(unsigned int, f);
    unsigned int r = u + 0x7FFFu + ((u >> 16) & 1u);   // round-to-nearest-even
    return (unsigned short)(r >> 16);
}
__device__ inline float sigmoidf_(float x) { return 1.f / (1.f + __expf(-x)); }

#define HAVE_TDM __has_builtin(__builtin_amdgcn_tensor_load_to_lds)

#if HAVE_TDM
// Build a 2-D TDM descriptor per CDNA5 ISA and issue
// TENSOR_LOAD_TO_LDS (6-arg builtin on this toolchain: g0,g1,g2,g3,g4,cpol).
// Tile = tile_d0 (contiguous elems) x tile_d1 (rows), fp32.
// Rows beyond tensor_d1 / elems beyond tensor_d0 are zero-filled by hardware (OOB rule).
__device__ inline void tdm_load_2d_f32(const float* gptr, unsigned lds_byte,
                                       unsigned tensor_d0, unsigned tensor_d1,
                                       unsigned tile_d0, unsigned tile_d1,
                                       unsigned long long row_stride_elems)
{
    unsigned long long ga = (unsigned long long)(uintptr_t)gptr;
    u32x4 g0;
    g0[0] = 1u;                                                  // count=1, user D#
    g0[1] = lds_byte;                                            // lds_addr (bytes)
    g0[2] = (unsigned)(ga & 0xFFFFFFFFu);                        // global_addr[31:0]
    g0[3] = (unsigned)((ga >> 32) & 0x01FFFFFFu) | (2u << 30);   // addr[56:32] | type=2

    i32x8 g1;
    g1[0] = (int)(2u << 16);                                     // data_size=2 (4B); mask/flags=0
    g1[1] = (int)((tensor_d0 & 0xFFFFu) << 16);                  // tensor_dim0[15:0] @ bits 63:48
    g1[2] = (int)((tensor_d0 >> 16) | ((tensor_d1 & 0xFFFFu) << 16)); // dim0 hi | dim1 lo
    g1[3] = (int)((tensor_d1 >> 16) | (tile_d0 << 16));          // dim1 hi | tile_dim0
    g1[4] = (int)(tile_d1 & 0xFFFFu);                            // tile_dim1 | tile_dim2=0
    g1[5] = (int)(row_stride_elems & 0xFFFFFFFFull);             // tensor_dim0_stride[31:0]
    g1[6] = (int)((row_stride_elems >> 32) & 0xFFFFull);         // stride[47:32] | dim1_stride lo=0
    g1[7] = 0;

    i32x4 z4 = {0, 0, 0, 0};
    i32x8 z8 = {0, 0, 0, 0, 0, 0, 0, 0};
    __builtin_amdgcn_tensor_load_to_lds(g0, g1, z4, z4, z8, 0);
}
__device__ inline unsigned lds_off(const void* p) {
    return (unsigned)(unsigned long long)(uintptr_t)p;            // low 32 bits = LDS byte offset
}
#endif

// ---------------- bf16 WMMA GEMM: C[M,N] = A[M,K] * W[N,K]^T (+bias[n]) ----------------
// block tile 128x64, 256 threads = 8 waves; wave w -> rows [w*16, w*16+16), 4 n-subtiles.
// Tiles staged global->LDS(fp32) by the Tensor Data Mover (double-buffered, wave 0
// produces, TENSORcnt synchronizes), then all waves convert to the padded bf16 tiles.
__global__ __launch_bounds__(256)
void gemm_bf16_wmma(const float* __restrict__ A, const float* __restrict__ W,
                    const float* __restrict__ bias, float* __restrict__ C,
                    int M, int N, int K)
{
#if HAVE_TDM
    __shared__ __align__(16) float Asf[2][128 * 32];
    __shared__ __align__(16) float Wsf[2][64 * 32];
#endif
    __shared__ __align__(16) unsigned short As[128 * 40];
    __shared__ __align__(16) unsigned short Ws[64 * 40];

    const int tid  = threadIdx.x;
    const int lane = tid & 31;
    const int w    = tid >> 5;
    const int gm0  = blockIdx.y * 128;
    const int gn0  = blockIdx.x * 64;
    const int ntiles = K / 32;

    v8f acc[4] = {};

#if HAVE_TDM
    // prologue: DMA tile 0 into buffer 0 (wave 0 only; TDM ignores EXEC, one op per wave)
    if (w == 0) {
        tdm_load_2d_f32(&A[(size_t)gm0 * K], lds_off(&Asf[0][0]),
                        (unsigned)K, 128u, 32u, 128u, (unsigned long long)K);
        tdm_load_2d_f32(&W[(size_t)gn0 * K], lds_off(&Wsf[0][0]),
                        (unsigned)K, (unsigned)(N - gn0), 32u, 64u, (unsigned long long)K);
    }
#endif

    for (int ki = 0; ki < ntiles; ++ki) {
        const int k0 = ki * 32;
#if HAVE_TDM
        if (w == 0) {
            if (ki + 1 < ntiles) {   // DMA next tile into the other buffer, overlap with WMMA
                int kn = k0 + 32;
                int nb = (ki + 1) & 1;
                tdm_load_2d_f32(&A[(size_t)gm0 * K + kn], lds_off(&Asf[nb][0]),
                                (unsigned)(K - kn), 128u, 32u, 128u, (unsigned long long)K);
                tdm_load_2d_f32(&W[(size_t)gn0 * K + kn], lds_off(&Wsf[nb][0]),
                                (unsigned)(K - kn), (unsigned)(N - gn0), 32u, 64u,
                                (unsigned long long)K);
                __builtin_amdgcn_s_wait_tensorcnt(2);   // in-order: tile ki complete
            } else {
                __builtin_amdgcn_s_wait_tensorcnt(0);
            }
        }
        __syncthreads();

        // convert staged fp32 tiles -> padded bf16 tiles (all 256 threads)
        const float* Af = &Asf[ki & 1][0];
        const float* Wf = &Wsf[ki & 1][0];
        #pragma unroll
        for (int it = 0; it < 4; ++it) {
            int r  = (tid >> 3) + it * 32;
            int c4 = (tid & 7) * 4;
            F4 v = *(const F4*)&Af[r * 32 + c4];
            unsigned short* d = &As[r * 40 + c4];
            d[0] = f2bf(v.x); d[1] = f2bf(v.y); d[2] = f2bf(v.z); d[3] = f2bf(v.w);
        }
        #pragma unroll
        for (int it = 0; it < 2; ++it) {
            int r  = (tid >> 3) + it * 32;
            int c4 = (tid & 7) * 4;
            F4 v = *(const F4*)&Wf[r * 32 + c4];
            unsigned short* d = &Ws[r * 40 + c4];
            d[0] = f2bf(v.x); d[1] = f2bf(v.y); d[2] = f2bf(v.z); d[3] = f2bf(v.w);
        }
#else
        // fallback: direct global->LDS staging with fp32->bf16 conversion
        #pragma unroll
        for (int it = 0; it < 4; ++it) {
            int r  = (tid >> 3) + it * 32;
            int c4 = (tid & 7) * 4;
            F4 v = *(const F4*)&A[(size_t)(gm0 + r) * K + k0 + c4];
            unsigned short* d = &As[r * 40 + c4];
            d[0] = f2bf(v.x); d[1] = f2bf(v.y); d[2] = f2bf(v.z); d[3] = f2bf(v.w);
        }
        #pragma unroll
        for (int it = 0; it < 2; ++it) {
            int r  = (tid >> 3) + it * 32;
            int c4 = (tid & 7) * 4;
            int gn = gn0 + r;
            F4 v = {0.f, 0.f, 0.f, 0.f};
            if (gn < N) v = *(const F4*)&W[(size_t)gn * K + k0 + c4];
            unsigned short* d = &Ws[r * 40 + c4];
            d[0] = f2bf(v.x); d[1] = f2bf(v.y); d[2] = f2bf(v.z); d[3] = f2bf(v.w);
        }
#endif
        __syncthreads();

        const int kh = lane >> 4;   // K-half per ISA 16-bit operand layout
        const int rl = lane & 15;

        Frag fa;   // A 16x32: lane row = rl, K = e + (e>=8?8:0) + 8*kh
        fa.q[0] = *(const U4*)&As[(w * 16 + rl) * 40 + 8 * kh];
        fa.q[1] = *(const U4*)&As[(w * 16 + rl) * 40 + 16 + 8 * kh];

        #pragma unroll
        for (int nt = 0; nt < 4; ++nt) {
            Frag fb;  // B 32x16: lane col = rl, K = e + (e>=8?8:0) + 8*kh
            fb.q[0] = *(const U4*)&Ws[(nt * 16 + rl) * 40 + 8 * kh];
            fb.q[1] = *(const U4*)&Ws[(nt * 16 + rl) * 40 + 16 + 8 * kh];
            acc[nt] = __builtin_amdgcn_wmma_f32_16x16x32_bf16(
                          false, fa.v, false, fb.v, (short)0, acc[nt], false, false);
        }
        __syncthreads();
    }

    // C/D layout: lane -> n = lane&15, rows m = r + 8*(lane>>4)
    const int rl = lane & 15;
    const int mh = (lane >> 4) * 8;
    #pragma unroll
    for (int nt = 0; nt < 4; ++nt) {
        int gn = gn0 + nt * 16 + rl;
        if (gn >= N) continue;
        float badd = bias ? bias[gn] : 0.f;
        #pragma unroll
        for (int r = 0; r < 8; ++r) {
            int gm = gm0 + w * 16 + mh + r;
            C[(size_t)gm * N + gn] = acc[nt][r] + badd;
        }
    }
    (void)M;
}

// ---------------- causal depthwise conv (k=4) + silu ----------------
__global__ __launch_bounds__(256)
void conv_silu_kernel(const float* __restrict__ zx, const float* __restrict__ cw,
                      const float* __restrict__ cb, float* __restrict__ co)
{
    size_t gid = (size_t)blockIdx.x * 256 + threadIdx.x;
    const size_t total = (size_t)BATCH * SEQLEN * CONV_DIMC;
    if (gid >= total) return;
    int c = (int)(gid % CONV_DIMC);
    size_t bt = gid / CONV_DIMC;
    int t = (int)(bt % SEQLEN);
    int b = (int)(bt / SEQLEN);
    float acc = cb[c];
    #pragma unroll
    for (int j = 0; j < D_CONV; ++j) {
        int tt = t - (D_CONV - 1) + j;
        if (tt >= 0)
            acc += zx[((size_t)b * SEQLEN + tt) * D_IN_PROJ + D_INNER + c] * cw[c * D_CONV + j];
    }
    co[gid] = acc * sigmoidf_(acc);
}

// ---------------- selective scan: 1 thread per (head,p), state[16] in registers ----------------
__global__ __launch_bounds__(64)
void scan_kernel(const float* __restrict__ zx, const float* __restrict__ co,
                 const float* __restrict__ dtb, const float* __restrict__ Alog,
                 const float* __restrict__ Dp, float* __restrict__ yb)
{
    int b = blockIdx.x / NHEADS;
    int h = blockIdx.x % NHEADS;
    int p = threadIdx.x;

    float A     = -__expf(Alog[h]);
    float dbias = dtb[h];
    float Dv    = Dp[h];

    float hs[D_STATE];
    #pragma unroll
    for (int n = 0; n < D_STATE; ++n) hs[n] = 0.f;

    for (int t = 0; t < SEQLEN; ++t) {
        size_t row = (size_t)b * SEQLEN + t;
        const float* zr = zx + row * D_IN_PROJ;
        const float* cr = co + row * CONV_DIMC;
        if (t + 1 < SEQLEN)   // latency-critical serial loop: prefetch next timestep's B/C
            __builtin_prefetch(cr + CONV_DIMC + D_INNER, 0, 1);

        float dtr = zr[D_IN_PROJ - NHEADS + h] + dbias;       // lane-uniform
        float dt  = (dtr > 20.f) ? dtr : log1pf(__expf(dtr)); // softplus
        float dA  = __expf(dt * A);
        float xs  = cr[h * HEADDIM + p];
        float dtx = dt * xs;

        F4 bb[4], cc[4];
        #pragma unroll
        for (int q = 0; q < 4; ++q) {
            bb[q] = *(const F4*)(cr + D_INNER + 4 * q);
            cc[q] = *(const F4*)(cr + D_INNER + D_STATE + 4 * q);
        }
        const float* Bv = (const float*)bb;
        const float* Cv = (const float*)cc;

        float yv = 0.f;
        #pragma unroll
        for (int n = 0; n < D_STATE; ++n) {
            hs[n] = hs[n] * dA + dtx * Bv[n];
            yv += hs[n] * Cv[n];
        }
        yb[row * D_INNER + h * HEADDIM + p] = yv + Dv * xs;
    }
}

// ---------------- gated RMSNorm (in-place on y): y = y*silu(z) * rsqrt(mean(y^2)+eps) * nw ----------------
__global__ __launch_bounds__(256)
void gated_norm_kernel(float* __restrict__ yb, const float* __restrict__ zx,
                       const float* __restrict__ nw)
{
    __shared__ float red[256];
    int row = blockIdx.x;
    int tid = threadIdx.x;
    const float* zr = zx + (size_t)row * D_IN_PROJ;   // z = first D_INNER entries
    float* yr = yb + (size_t)row * D_INNER;

    float vals[6];
    float ss = 0.f;
    #pragma unroll
    for (int j = 0; j < 6; ++j) {
        int i = tid + j * 256;
        float z = zr[i];
        float v = yr[i] * (z * sigmoidf_(z));
        vals[j] = v;
        ss += v * v;
    }
    red[tid] = ss;
    __syncthreads();
    for (int s = 128; s > 0; s >>= 1) {
        if (tid < s) red[tid] += red[tid + s];
        __syncthreads();
    }
    float scale = rsqrtf(red[0] / (float)D_INNER + 1e-5f);
    #pragma unroll
    for (int j = 0; j < 6; ++j) {
        int i = tid + j * 256;
        yr[i] = vals[j] * scale * nw[i];
    }
}

// ---------------- head epilogue: sigmoid (except velocity) + type mask, write stacked output ----------------
__global__ __launch_bounds__(256)
void head_out_kernel(const float* __restrict__ lg0, const float* __restrict__ lg1,
                     const int* __restrict__ type_ids, float* __restrict__ out)
{
    size_t gid = (size_t)blockIdx.x * 256 + threadIdx.x;
    const size_t total = (size_t)8 * BATCH * SEQLEN * NPITCH;
    if (gid >= total) return;
    int p = (int)(gid % NPITCH);
    size_t r = gid / NPITCH;
    int l = (int)(r % SEQLEN); r /= SEQLEN;
    int b = (int)(r % BATCH);
    int s = (int)(r / BATCH);
    int head = s >> 2;
    int slot = s & 3;
    // stacked order per head: velocity(j0), frames(j3), onset(j1), offset(j2)
    const int jmap[4] = {0, 3, 1, 2};
    int j = jmap[slot];
    const float* lg = head ? lg1 : lg0;
    float v = lg[((size_t)b * SEQLEN + l) * (4 * NPITCH) + j * NPITCH + p];
    if (slot != 0) v = sigmoidf_(v);
    int ty = type_ids[(size_t)b * SEQLEN + l];
    out[gid] = (ty == head) ? v : 0.f;
}

// ---------------- driver ----------------
static inline dim3 gemm_grid(int M, int N) { return dim3((N + 63) / 64, M / 128, 1); }

extern "C" void kernel_launch(void* const* d_in, const int* in_sizes, int n_in,
                              void* d_out, int out_size, void* d_ws, size_t ws_size,
                              hipStream_t stream)
{
    (void)in_sizes; (void)n_in; (void)out_size; (void)ws_size;
    const float* x_in       = (const float*)d_in[0];
    const int*   type_ids   = (const int*)  d_in[1];
    const float* in_proj_w  = (const float*)d_in[2];
    const float* conv_w     = (const float*)d_in[3];
    const float* conv_b     = (const float*)d_in[4];
    const float* dt_bias    = (const float*)d_in[5];
    const float* A_log      = (const float*)d_in[6];
    const float* Dparam     = (const float*)d_in[7];
    const float* norm_w     = (const float*)d_in[8];
    const float* out_proj_w = (const float*)d_in[9];
    const float* head_w     = (const float*)d_in[10];
    const float* head_b     = (const float*)d_in[11];
    const float* head_out_w = (const float*)d_in[12];
    const float* head_out_b = (const float*)d_in[13];
    float* out = (float*)d_out;

    // workspace layout (floats)
    float* ws      = (float*)d_ws;
    float* zx      = ws;                                     // MROWS * D_IN_PROJ
    float* convout = zx      + (size_t)MROWS * D_IN_PROJ;    // MROWS * CONV_DIMC
    float* ybuf    = convout + (size_t)MROWS * CONV_DIMC;    // MROWS * D_INNER
    float* xA      = ybuf    + (size_t)MROWS * D_INNER;      // MROWS * DIMD
    float* xB      = xA      + (size_t)MROWS * DIMD;         // MROWS * DIMD
    float* hbuf    = xB      + (size_t)MROWS * DIMD;         // MROWS * DIMD
    float* lg0     = hbuf    + (size_t)MROWS * DIMD;         // MROWS * 512
    float* lg1     = lg0     + (size_t)MROWS * 512;          // MROWS * 512

    const float* cur = x_in;
    float* nxt = xA;

    const size_t conv_total = (size_t)BATCH * SEQLEN * CONV_DIMC;
    const int conv_blocks = (int)((conv_total + 255) / 256);

    for (int i = 0; i < BLOCKS; ++i) {
        // zxbcdt = x @ Wi^T
        gemm_bf16_wmma<<<gemm_grid(MROWS, D_IN_PROJ), 256, 0, stream>>>(
            cur, in_proj_w + (size_t)i * D_IN_PROJ * DIMD, nullptr, zx,
            MROWS, D_IN_PROJ, DIMD);
        // depthwise conv + silu
        conv_silu_kernel<<<conv_blocks, 256, 0, stream>>>(
            zx, conv_w + (size_t)i * CONV_DIMC * D_CONV,
            conv_b + (size_t)i * CONV_DIMC, convout);
        // selective scan
        scan_kernel<<<BATCH * NHEADS, 64, 0, stream>>>(
            zx, convout,
            dt_bias + (size_t)i * NHEADS, A_log + (size_t)i * NHEADS,
            Dparam + (size_t)i * NHEADS, ybuf);
        // gated RMSNorm (in place)
        gated_norm_kernel<<<MROWS, 256, 0, stream>>>(
            ybuf, zx, norm_w + (size_t)i * D_INNER);
        // x = y @ Wo^T
        gemm_bf16_wmma<<<gemm_grid(MROWS, DIMD), 256, 0, stream>>>(
            ybuf, out_proj_w + (size_t)i * DIMD * D_INNER, nullptr, nxt,
            MROWS, DIMD, D_INNER);
        cur = nxt;
        nxt = (nxt == xA) ? xB : xA;
    }

    // heads
    for (int head = 0; head < 2; ++head) {
        float* lg = head ? lg1 : lg0;
        gemm_bf16_wmma<<<gemm_grid(MROWS, DIMD), 256, 0, stream>>>(
            cur, head_w + (size_t)head * DIMD * DIMD,
            head_b + (size_t)head * DIMD, hbuf, MROWS, DIMD, DIMD);
        gemm_bf16_wmma<<<gemm_grid(MROWS, 4 * NPITCH), 256, 0, stream>>>(
            hbuf, head_out_w + (size_t)head * 4 * NPITCH * DIMD,
            head_out_b + (size_t)head * 4 * NPITCH, lg, MROWS, 4 * NPITCH, DIMD);
    }

    const size_t out_total = (size_t)8 * BATCH * SEQLEN * NPITCH;
    head_out_kernel<<<(int)((out_total + 255) / 256), 256, 0, stream>>>(
        lg0, lg1, type_ids, out);
}
